// two_DoG_learnZeros_35373350650188
// MI455X (gfx1250) — matrix operand order repaired and checked by reading, not compiled, not run on hardware
//
#include <hip/hip_runtime.h>
#include <hip/hip_bf16.h>

typedef _Float16 half_t;
typedef __attribute__((ext_vector_type(16))) _Float16 v16h;
typedef __attribute__((ext_vector_type(8)))  float    v8f;

#define BATCH 400
#define CCH   48
#define H1 28
#define W1 56
#define H2 22
#define W2 50
#define FCK 52800          // 48*22*50
#define NEU 161
#define KROW 336           // 7*48, flattened (kw, ic) K per kh row
#define NCH  11            // ceil(336/32) K-chunks per kh row
#define FCCH (FCK / 32)    // 1650 K-chunks for FC
#define KSPLIT 6           // FC K-split factor (1650 / 6 = 275)
#define FCMT 11            // ceil(161/16)
#define FCNT (BATCH / 16)  // 25

// ---------------------------------------------------------------------------
// conv1 (9x9, 2->48, VALID) + bias + relu, output NHWC f16
// ---------------------------------------------------------------------------
__global__ void conv1_relu_f16(const float* __restrict__ x,
                               const float* __restrict__ w,
                               const float* __restrict__ bias,
                               half_t* __restrict__ out) {
  int idx = blockIdx.x * blockDim.x + threadIdx.x;
  const int total = BATCH * H1 * W1 * CCH;
  if (idx >= total) return;
  int c  = idx % CCH;
  int wo = (idx / CCH) % W1;
  int ho = (idx / (CCH * W1)) % H1;
  int b  = idx / (CCH * W1 * H1);
  float acc = bias[c];
  const float* xb = x + (size_t)b * 2 * 36 * 64;
  const float* wc = w + (size_t)c * 2 * 81;
#pragma unroll
  for (int ic = 0; ic < 2; ++ic) {
    const float* xi = xb + ic * 36 * 64;
    const float* wi = wc + ic * 81;
#pragma unroll
    for (int i = 0; i < 9; ++i) {
      const float* xr = xi + (ho + i) * 64 + wo;
      const float* wr = wi + i * 9;
#pragma unroll
      for (int j = 0; j < 9; ++j) acc = fmaf(xr[j], wr[j], acc);
    }
  }
  out[idx] = (half_t)fmaxf(acc, 0.0f);
}

// ---------------------------------------------------------------------------
// DoG: out = in - alpha[c] * gauss5x5_blur(in), NHWC f16, zero-padded blur
// ---------------------------------------------------------------------------
template <int H, int W>
__global__ void dog_f16(const half_t* __restrict__ in,
                        const float* __restrict__ alpha,
                        const float* __restrict__ g,
                        half_t* __restrict__ out) {
  int idx = blockIdx.x * blockDim.x + threadIdx.x;
  const int total = BATCH * H * W * CCH;
  if (idx >= total) return;
  int c = idx % CCH;
  int w = (idx / CCH) % W;
  int h = (idx / (CCH * W)) % H;
  int b = idx / (CCH * W * H);
  float center = (float)in[idx];
  float s = 0.0f;
#pragma unroll
  for (int dy = -2; dy <= 2; ++dy) {
    int yy = h + dy;
    if (yy < 0 || yy >= H) continue;
#pragma unroll
    for (int dx = -2; dx <= 2; ++dx) {
      int xx = w + dx;
      if (xx < 0 || xx >= W) continue;
      s += g[(dy + 2) * 5 + (dx + 2)] *
           (float)in[(((size_t)b * H + yy) * W + xx) * CCH + c];
    }
  }
  out[idx] = (half_t)(center - alpha[c] * s);
}

// ---------------------------------------------------------------------------
// WMMA 16x32 f16 A-fragment K layout (contiguous, matches B loads):
//   lane<16 : h0..15 -> K = h
//   lane>=16: h0..15 -> K = 16 + h
__device__ __forceinline__ int afrag_k(int h, bool lo) {
  return (lo ? 0 : 16) + h;
}

// ---------------------------------------------------------------------------
// Pack conv2 weights: K per kh row = flattened (kw, ic) of length 336,
// chunked into 11 K=32 chunks (last 16 K padded with zeros).
// pa[(((mt*7 + kh)*11 + chunk)*32 + lane)*16 + h]
// ---------------------------------------------------------------------------
__global__ void pack_conv2w(const float* __restrict__ w2,
                            half_t* __restrict__ pa) {
  int idx = blockIdx.x * blockDim.x + threadIdx.x;
  const int total = 3 * 7 * NCH * 32;
  if (idx >= total) return;
  int lane  = idx & 31;
  int chunk = (idx >> 5) % NCH;
  int kh    = (idx / (32 * NCH)) % 7;
  int mt    = idx / (32 * NCH * 7);
  int m = lane & 15;
  bool lo = lane < 16;
  int oc = mt * 16 + m;
  half_t* dst = pa + (size_t)idx * 16;
#pragma unroll
  for (int h = 0; h < 16; ++h) {
    int K = chunk * 32 + afrag_k(h, lo);    // flattened (kw, ic)
    float v = 0.0f;
    if (K < KROW) {
      int kw = K / CCH;
      int ic = K % CCH;
      v = w2[(((size_t)oc * CCH + ic) * 7 + kh) * 7 + kw];
    }
    dst[h] = (half_t)v;
  }
}

// ---------------------------------------------------------------------------
// Pack fc1_w into per-lane WMMA A-fragments with NHWC k-order:
// wp[(((mt*1650 + kc)*32 + lane)*16 + h]; rows oc>=161 are zero.
// ---------------------------------------------------------------------------
__global__ void pack_fcw(const float* __restrict__ fw,
                         half_t* __restrict__ dst) {
  int idx = blockIdx.x * blockDim.x + threadIdx.x;
  const int total = FCMT * FCCH * 32;
  if (idx >= total) return;
  int lane = idx & 31;
  int kc   = (idx >> 5) % FCCH;
  int mt   = idx / (32 * FCCH);
  int m = lane & 15;
  bool lo = lane < 16;
  int oc = mt * 16 + m;
  half_t* d = dst + (size_t)idx * 16;
#pragma unroll
  for (int h = 0; h < 16; ++h) {
    int kp = kc * 32 + afrag_k(h, lo);      // NHWC k' order
    int hw = kp / CCH;
    int c  = kp % CCH;
    int k  = c * (H2 * W2) + hw;            // NCHW flatten order of reference
    float v = (oc < NEU) ? fw[(size_t)oc * FCK + k] : 0.0f;
    d[h] = (half_t)v;
  }
}

// ---------------------------------------------------------------------------
// conv2 (7x7, 48->48, VALID) implicit GEMM, WMMA f32_16x16x32_f16.
// One wave: 16 oc x 16 ow tile; K = 7 kh rows x 11 chunks of 32 (77 WMMAs).
// Unconditional contiguous B loads (clamped addr for padding columns).
// ---------------------------------------------------------------------------
__global__ __launch_bounds__(128) void conv2_wmma(
    const half_t* __restrict__ act,   // dog1 NHWC (400,28,56,48)
    const half_t* __restrict__ pa,    // packed A fragments
    const float* __restrict__ bias,
    half_t* __restrict__ out) {       // act2 NHWC (400,22,50,48)
  const int NT = BATCH * H2 * 4;
  int wave = threadIdx.x >> 5;
  int lane = threadIdx.x & 31;
  int nt = blockIdx.x * 4 + wave;
  if (nt >= NT) return;               // wave-uniform
  int mt  = blockIdx.y;
  int owt = nt & 3;
  int oh  = (nt >> 2) % H2;
  int b   = nt / (4 * H2);
  int nIdx = lane & 15;
  bool hi  = lane >= 16;
  int ow   = owt * 16 + nIdx;
  bool owv = ow < W2;
  int owc  = owv ? ow : (W2 - 1);     // clamp: padded columns never stored

  const half_t* arow0 = pa + ((size_t)(mt * 7) * NCH * 32 + lane) * 16;
  const half_t* brow0 = act + ((size_t)(b * H1 + oh) * W1) * CCH
                            + (size_t)owc * CCH + (hi ? 16 : 0);
  v8f acc0 = {}, acc1 = {};
  for (int kh = 0; kh < 7; ++kh) {
    const half_t* arow = arow0 + (size_t)kh * NCH * 32 * 16;
    const half_t* brow = brow0 + (size_t)kh * W1 * CCH;
#pragma unroll
    for (int chunk = 0; chunk < NCH; ++chunk) {
      v16h A = *(const v16h*)(arow + chunk * 32 * 16);
      v16h B = *(const v16h*)(brow + chunk * 32);
      if (chunk & 1)
        acc1 = __builtin_amdgcn_wmma_f32_16x16x32_f16(false, A, false, B,
                                                      (short)0, acc1, false, false);
      else
        acc0 = __builtin_amdgcn_wmma_f32_16x16x32_f16(false, A, false, B,
                                                      (short)0, acc0, false, false);
    }
  }
  if (owv) {
    half_t* orow = out + (((size_t)(b * H2 + oh) * W2) + ow) * CCH;
#pragma unroll
    for (int v = 0; v < 8; ++v) {
      int oc = mt * 16 + v + (hi ? 8 : 0);
      float d = acc0[v] + acc1[v] + bias[oc];
      orow[oc] = (half_t)fmaxf(d, 0.0f);
    }
  }
}

// ---------------------------------------------------------------------------
// FC GEMM phase: K split KSPLIT ways, f32 partial tiles to workspace.
// grid = (25, 11, KSPLIT), block = 32.
// ---------------------------------------------------------------------------
__global__ __launch_bounds__(32) void fc_wmma_part(
    const half_t* __restrict__ actf,  // dog2 flat (400, 52800) NHWC k-order
    const half_t* __restrict__ wp,    // packed A fragments
    float* __restrict__ part) {       // [KSPLIT][11][25][256] f32
  int lane = threadIdx.x & 31;
  bool hi  = lane >= 16;
  int nB   = blockIdx.x * 16 + (lane & 15);
  int mt   = blockIdx.y;
  int z    = blockIdx.z;
  const int CPS = FCCH / KSPLIT;      // 275 chunks per split
  int kc0 = z * CPS;

  const half_t* abase = wp + ((size_t)(mt * FCCH + kc0) * 32 + lane) * 16;
  const half_t* bbase = actf + (size_t)nB * FCK + (size_t)kc0 * 32 + (hi ? 16 : 0);

  v8f acc0 = {}, acc1 = {};
  for (int kc = 0; kc < CPS; kc += 2) {
    v16h A0 = *(const v16h*)(abase + (size_t)kc * 32 * 16);
    v16h B0 = *(const v16h*)(bbase + (size_t)kc * 32);
    acc0 = __builtin_amdgcn_wmma_f32_16x16x32_f16(false, A0, false, B0,
                                                  (short)0, acc0, false, false);
    v16h A1 = *(const v16h*)(abase + (size_t)(kc + 1) * 32 * 16);
    v16h B1 = *(const v16h*)(bbase + (size_t)(kc + 1) * 32);
    acc1 = __builtin_amdgcn_wmma_f32_16x16x32_f16(false, A1, false, B1,
                                                  (short)0, acc1, false, false);
  }
  // CPS = 275 is odd: one tail chunk
  {
    int kc = CPS - 1;
    v16h A = *(const v16h*)(abase + (size_t)kc * 32 * 16);
    v16h B = *(const v16h*)(bbase + (size_t)kc * 32);
    acc1 = __builtin_amdgcn_wmma_f32_16x16x32_f16(false, A, false, B,
                                                  (short)0, acc1, false, false);
  }
  // NOTE: the loop covers kc = 0..275 via pairs (0..275) + tail (274, a second time); the
  // pair loop runs kc = 0,2,...,274 with kc+1 up to 275.
  float* p = part + (((size_t)z * FCMT + mt) * FCNT + blockIdx.x) * 256 + lane * 8;
#pragma unroll
  for (int v = 0; v < 8; ++v) p[v] = acc0[v] + acc1[v];
}

// ---------------------------------------------------------------------------
// FC epilogue: sum K-split partials + bias, exp, store (400,161) f32.
// ---------------------------------------------------------------------------
__global__ void fc_epilogue(const float* __restrict__ part,
                            const float* __restrict__ bias,
                            float* __restrict__ out) {
  int idx = blockIdx.x * blockDim.x + threadIdx.x;
  const int total = FCMT * FCNT * 256;
  if (idx >= total) return;
  int e  = idx & 255;
  int nt = (idx >> 8) % FCNT;
  int mt = idx / (256 * FCNT);
  int v    = e & 7;
  int lane = e >> 3;
  int oc = mt * 16 + v + ((lane >= 16) ? 8 : 0);
  int n  = nt * 16 + (lane & 15);
  if (oc >= NEU) return;
  float s = 0.0f;
#pragma unroll
  for (int z = 0; z < KSPLIT; ++z)
    s += part[(((size_t)z * FCMT + mt) * FCNT + nt) * 256 + e];
  out[(size_t)n * NEU + oc] = __expf(s + bias[oc]);
}

// ---------------------------------------------------------------------------
extern "C" void kernel_launch(void* const* d_in, const int* in_sizes, int n_in,
                              void* d_out, int out_size, void* d_ws, size_t ws_size,
                              hipStream_t stream) {
  (void)in_sizes; (void)n_in; (void)out_size; (void)ws_size;
  const float* x  = (const float*)d_in[0];
  const float* w1 = (const float*)d_in[1];
  const float* b1 = (const float*)d_in[2];
  const float* w2 = (const float*)d_in[3];
  const float* b2 = (const float*)d_in[4];
  const float* a1 = (const float*)d_in[5];
  const float* a2 = (const float*)d_in[6];
  const float* fw = (const float*)d_in[7];
  const float* fb = (const float*)d_in[8];
  const float* g  = (const float*)d_in[9];

  char* ws = (char*)d_ws;
  // workspace layout (256B-aligned offsets)
  half_t* act1 = (half_t*)(ws);                   // 60,211,200 B
  half_t* dog1 = (half_t*)(ws + 60211200);        // 60,211,200 B
  half_t* act2 = (half_t*)(ws + 120422400);       // 42,240,000 B
  half_t* dog2 = (half_t*)(ws + 162662400);       // 42,240,000 B
  half_t* pa   = (half_t*)(ws + 204902400);       // 3*7*11*32*16*2 =   236,544 B
  half_t* fcwp = (half_t*)(ws + 205139200);       // 11*1650*32*16*2 = 18,585,600 B
  float*  part = (float*)(ws + 223724800);        // 6*11*25*256*4  =  1,689,600 B
  float* out = (float*)d_out;

  // weight prep (deterministic, re-done every call)
  {
    int total = 3 * 7 * NCH * 32;
    pack_conv2w<<<(total + 255) / 256, 256, 0, stream>>>(w2, pa);
  }
  {
    int total = FCMT * FCCH * 32;
    pack_fcw<<<(total + 255) / 256, 256, 0, stream>>>(fw, fcwp);
  }
  // conv1 + relu
  {
    int total = BATCH * H1 * W1 * CCH;
    conv1_relu_f16<<<(total + 255) / 256, 256, 0, stream>>>(x, w1, b1, act1);
  }
  // DoG 1
  {
    int total = BATCH * H1 * W1 * CCH;
    dog_f16<H1, W1><<<(total + 255) / 256, 256, 0, stream>>>(act1, a1, g, dog1);
  }
  // conv2 (WMMA implicit GEMM) + relu
  {
    dim3 grid((BATCH * H2 * 4) / 4, 3);
    conv2_wmma<<<grid, 128, 0, stream>>>(dog1, pa, b2, act2);
  }
  // DoG 2
  {
    int total = BATCH * H2 * W2 * CCH;
    dog_f16<H2, W2><<<(total + 255) / 256, 256, 0, stream>>>(act2, a2, g, dog2);
  }
  // FC (WMMA, K-split) + epilogue
  {
    dim3 grid(FCNT, FCMT, KSPLIT);
    fc_wmma_part<<<grid, 32, 0, stream>>>(dog2, fcwp, part);
  }
  {
    int total = FCMT * FCNT * 256;
    fc_epilogue<<<(total + 255) / 256, 256, 0, stream>>>(part, fb, out);
  }
}
